// VarianceARPredictor_7456063226510
// MI455X (gfx1250) — matrix operand order repaired and checked by reading, not compiled, not run on hardware
//
#include <hip/hip_runtime.h>
#include <hip/hip_fp16.h>

typedef _Float16 half_t;
typedef __attribute__((ext_vector_type(16))) _Float16 v16h;
typedef __attribute__((ext_vector_type(8)))  _Float16 v8h;
typedef __attribute__((ext_vector_type(8)))  float    v8f;
typedef __attribute__((ext_vector_type(4)))  unsigned int u32x4;
typedef __attribute__((ext_vector_type(8)))  int          i32x8;
typedef __attribute__((ext_vector_type(4)))  int          i32x4;

#define DEV static __device__ __forceinline__

// ---------------------------------------------------------------- fragments
DEV v16h frag_join(v8h lo, v8h hi) {
    v16h r;
#pragma unroll
    for (int i = 0; i < 8; ++i) { r[i] = lo[i]; r[i + 8] = hi[i]; }
    return r;
}

// A-matrix 16x32 f16 (ISA 7.12.2): lanes 0-15 hold K{kk..kk+7, kk+16..kk+23},
// lanes 16-31 hold K{kk+8..kk+15, kk+24..kk+31}. sel = lane>>4.
DEV v16h load_afrag(const half_t* base, int stride, int row, int kk, int sel) {
    const half_t* p = base + (size_t)row * stride + kk + sel * 8;
    v8h lo = *(const v8h*)(p);
    v8h hi = *(const v8h*)(p + 16);
    return frag_join(lo, hi);
}

// B-matrix 32x16 f16: lane n holds column n, same K order as A
// (lanes 0-15: K{kk..kk+7, kk+16..kk+23}, lanes 16-31: K{kk+8..kk+15, kk+24..kk+31}). Column n of
// B == row n of the row-major weight array (gates = x @ W^T).
DEV v16h load_bfrag(const half_t* base, int stride, int col, int kk, int sel) {
    const half_t* p = base + (size_t)col * stride + kk + sel * 8;
    v8h lo = *(const v8h*)(p);
    v8h hi = *(const v8h*)(p + 16);
    return frag_join(lo, hi);
}

DEV v8f wmma16(v16h a, v16h b, v8f c) {
    return __builtin_amdgcn_wmma_f32_16x16x32_f16(false, a, false, b,
                                                  (short)0, c, false, false);
}

DEV v16h zero16() { v16h z; for (int i = 0; i < 16; ++i) z[i] = (half_t)0; return z; }
DEV v8f  zero8()  { v8f  z; for (int i = 0; i < 8;  ++i) z[i] = 0.f;       return z; }
DEV float sigm(float x) { return 1.f / (1.f + __expf(-x)); }

// ---------------------------------------------------------------- TDM staging
// One D# describing a 3D tile: x = xlen contiguous halves, y = 32 rows with
// stride `rstride` halves, z = 4 gate blocks with stride `gstride` halves.
// Moves the whole [4][32][xlen] f16 slab global->LDS in a single DMA op,
// tracked by TENSORcnt (ISA ch.8: Tensor Data Mover). This toolchain exposes
// the 6-arg builtin (g0, g1, g2, g3, g4, cpol); trailing group is unused here.
DEV void tdm_load_slab(const half_t* gsrc, half_t* ldst, unsigned xlen,
                       unsigned rstride, unsigned gstride) {
    unsigned lds = (unsigned)(uintptr_t)ldst;                 // LDS byte offset
    unsigned long long ga = (unsigned long long)(uintptr_t)gsrc;
    u32x4 g0;
    g0[0] = 1u;                                    // count=1 (valid user D#)
    g0[1] = lds;                                   // lds_addr [63:32]
    g0[2] = (unsigned)(ga & 0xffffffffu);          // global_addr [95:64]
    g0[3] = (unsigned)((ga >> 32) & 0x1ffffffu)    // global_addr [120:96]
          | (2u << 30);                            // type=2 ("image") [127:126]
    i32x8 g1;
    g1[0] = (1 << 16);                             // data_size=1 -> 2 bytes
    g1[1] = (int)((xlen & 0xffffu) << 16);         // tensor_dim0[15:0] @ [63:48]
    g1[2] = (int)(((xlen >> 16) & 0xffffu)         // tensor_dim0[31:16] @ [79:64]
          | (32u << 16));                          // tensor_dim1[15:0]=32 @ [95:80]
    g1[3] = (int)((xlen & 0xffffu) << 16);         // tile_dim0 @ [127:112]
    g1[4] = (int)(32u | (4u << 16));               // tile_dim1=32, tile_dim2=4
    g1[5] = (int)rstride;                          // tensor_dim0_stride[31:0]
    g1[6] = (int)((gstride & 0xffffu) << 16);      // dim1_stride[15:0] @ [223:208]
    g1[7] = (int)(gstride >> 16);                  // dim1_stride[47:16]
    i32x4 g2;
    g2[0] = 4; g2[1] = 0; g2[2] = 0; g2[3] = 0;    // tensor_dim2=4
    i32x4 g3;
    g3[0] = 0; g3[1] = 0; g3[2] = 0; g3[3] = 0;
    i32x8 g4;
#pragma unroll
    for (int i = 0; i < 8; ++i) g4[i] = 0;         // unused trailing group
    __builtin_amdgcn_tensor_load_to_lds(g0, g1, g2, g3, g4, 0);
}

// ---------------------------------------------------------------- prep kernels
__global__ void k_f32_to_f16(const float* __restrict__ s, half_t* __restrict__ d, int n) {
    int i = blockIdx.x * blockDim.x + threadIdx.x;
    if (i < n) d[i] = (half_t)s[i];
}

__global__ void k_bias_sum(const float* a, const float* b, float* o, int n) {
    int i = blockIdx.x * blockDim.x + threadIdx.x;
    if (i < n) o[i] = a[i] + b[i];
}

__global__ void k_zero_u32(unsigned* p, int n) {
    int i = blockIdx.x * blockDim.x + threadIdx.x;
    if (i < n) p[i] = 0u;
}

// conv w [Cout][Cin][3] -> wt[n][dt*Cin + c] (f16, B-friendly row-major in K)
__global__ void k_repack_convw(const float* __restrict__ w, half_t* __restrict__ wt,
                               int Cout, int Cin) {
    int i = blockIdx.x * blockDim.x + threadIdx.x;
    int total = Cout * Cin * 3;
    if (i < total) {
        int n = i / (Cin * 3), r = i % (Cin * 3);
        int c = r / 3, dt = r % 3;
        wt[(size_t)n * (3 * Cin) + dt * Cin + c] = (half_t)w[i];
    }
}

// prenet half of xs: xs[(t*32+b)][512+f] = prev*pw[f]+pb[f]
__global__ void k_prenet(const float* __restrict__ target, const float* __restrict__ pw,
                         const float* __restrict__ pb, half_t* __restrict__ xs) {
    int i = blockIdx.x * blockDim.x + threadIdx.x;
    if (i < 32000 * 512) {
        int f = i & 511, row = i >> 9;       // row = t*32+b
        int t = row >> 5, b = row & 31;
        float prev = (t == 0) ? 0.f : target[b * 1000 + (t - 1)];
        xs[(size_t)row * 1024 + 512 + f] = (half_t)(prev * pw[f] + pb[f]);
    }
}

// ---------------------------------------------------------------- conv as WMMA GEMM
// y[bt][n] = relu( sum_{dt,c} x[b, t+dt-1, c] * wt[n][dt*Cin+c] + bias[n] )
__global__ __launch_bounds__(256) void k_conv_wmma(const half_t* __restrict__ xin,
                                                   const half_t* __restrict__ wt,
                                                   const float* __restrict__ bias,
                                                   float* __restrict__ y, int Cin) {
    const int lane = threadIdx.x & 31, wv = threadIdx.x >> 5;
    const int sel = lane >> 4, l15 = lane & 15;
    const int mt = blockIdx.x;                 // 0..1999
    const int arow = mt * 16 + l15;            // bt index this lane feeds A with
    const int b = arow / 1000, t = arow - b * 1000;
    const int K = 3 * Cin, nb0 = wv * 64;
    v8f acc[4]; for (int s = 0; s < 4; ++s) acc[s] = zero8();
    for (int kk = 0; kk < K; kk += 32) {
        int dt = kk / Cin, c = kk - dt * Cin, tt = t + dt - 1;
        v16h a = (tt >= 0 && tt < 1000) ? load_afrag(xin, Cin, b * 1000 + tt, c, sel)
                                        : zero16();
#pragma unroll
        for (int s = 0; s < 4; ++s) {
            v16h bf = load_bfrag(wt, K, nb0 + s * 16 + l15, kk, sel);
            acc[s] = wmma16(a, bf, acc[s]);
        }
    }
#pragma unroll
    for (int s = 0; s < 4; ++s) {
        int n = nb0 + s * 16 + l15;
        float bv = bias[n];
#pragma unroll
        for (int r = 0; r < 8; ++r) {
            int m = mt * 16 + sel * 8 + r;
            float v = acc[s][r] + bv;
            y[(size_t)m * 512 + n] = v > 0.f ? v : 0.f;
        }
    }
}

// ---------------------------------------------------------------- LayerNorm (post-relu)
template <int MODE>
__global__ __launch_bounds__(256) void k_ln(const float* __restrict__ y,
                                            const float* __restrict__ g,
                                            const float* __restrict__ bb,
                                            half_t* __restrict__ out_h,
                                            float* __restrict__ enc_f,
                                            half_t* __restrict__ xs) {
    const int wv = threadIdx.x >> 5, lane = threadIdx.x & 31;
    const int row = blockIdx.x * 8 + wv;       // b*1000+t
    float v[16]; float s = 0.f;
#pragma unroll
    for (int j = 0; j < 16; ++j) {
        float x = y[(size_t)row * 512 + lane + j * 32];
        x = x > 0.f ? x : 0.f;                 // relu happens before LN stats
        v[j] = x; s += x;
    }
    for (int m = 1; m < 32; m <<= 1) s += __shfl_xor(s, m, 32);
    float mu = s * (1.f / 512.f), var = 0.f;
#pragma unroll
    for (int j = 0; j < 16; ++j) { float d = v[j] - mu; var += d * d; }
    for (int m = 1; m < 32; m <<= 1) var += __shfl_xor(var, m, 32);
    float inv = rsqrtf(var * (1.f / 512.f) + 1e-5f);
#pragma unroll
    for (int j = 0; j < 16; ++j) {
        int c = lane + j * 32;
        float r = (v[j] - mu) * inv * g[c] + bb[c];
        if (MODE == 0) {
            out_h[(size_t)row * 512 + c] = (half_t)r;
        } else {
            enc_f[(size_t)row * 512 + c] = r;
            int t = row % 1000, b = row / 1000;
            xs[((size_t)(t * 32 + b)) * 1024 + c] = (half_t)r;
        }
    }
}

// ---------------------------------------------------------------- gates1x = xs @ w_ih1^T + b1
__global__ __launch_bounds__(256) void k_gemm_ih1(const half_t* __restrict__ xs,
                                                  const half_t* __restrict__ wih1,
                                                  const float* __restrict__ b1,
                                                  float* __restrict__ g1x) {
    const int lane = threadIdx.x & 31, wv = threadIdx.x >> 5;
    const int sel = lane >> 4, l15 = lane & 15;
    const int mt = blockIdx.x;
    const int nb0 = blockIdx.y * 512 + wv * 64;
    v8f acc[4]; for (int s = 0; s < 4; ++s) acc[s] = zero8();
    for (int kk = 0; kk < 1024; kk += 32) {
        __builtin_prefetch(xs + ((size_t)(mt * 16 + l15)) * 1024 + kk + 128, 0, 1);
        v16h a = load_afrag(xs, 1024, mt * 16 + l15, kk, sel);
#pragma unroll
        for (int s = 0; s < 4; ++s) {
            v16h bf = load_bfrag(wih1, 1024, nb0 + s * 16 + l15, kk, sel);
            acc[s] = wmma16(a, bf, acc[s]);
        }
    }
#pragma unroll
    for (int s = 0; s < 4; ++s) {
        int n = nb0 + s * 16 + l15;
        float bv = b1[n];
#pragma unroll
        for (int r = 0; r < 8; ++r) {
            int m = mt * 16 + sel * 8 + r;
            g1x[(size_t)m * 2048 + n] = acc[s][r] + bv;
        }
    }
}

// ---------------------------------------------------------------- grid barrier
DEV void grid_barrier(unsigned* bar, unsigned nwg, int tid) {
    __syncthreads();
    if (tid == 0) {
        __threadfence();
        unsigned gen = __hip_atomic_load(&bar[1], __ATOMIC_ACQUIRE, __HIP_MEMORY_SCOPE_AGENT);
        unsigned arrived = atomicAdd(&bar[0], 1u);
        if (arrived == nwg - 1) {
            __hip_atomic_store(&bar[0], 0u, __ATOMIC_RELAXED, __HIP_MEMORY_SCOPE_AGENT);
            __hip_atomic_fetch_add(&bar[1], 1u, __ATOMIC_RELEASE, __HIP_MEMORY_SCOPE_AGENT);
        } else {
            while (__hip_atomic_load(&bar[1], __ATOMIC_ACQUIRE, __HIP_MEMORY_SCOPE_AGENT) == gen)
                __builtin_amdgcn_s_sleep(1);
        }
    }
    __syncthreads();
}

// ---------------------------------------------------------------- persistent LSTM
// 16 WGs x 128 threads. WG g owns hidden slice [g*32, g*32+32).
// Wave (mt,nblk) owns batch tile mt (16 rows) x hidden 16-block nblk, all 4 gates.
// w_hh1/w_hh2 slices are DMA'd into LDS once by the TDM and reused 1000x;
// c-state persists in VGPRs for the whole sequence.
__global__ __launch_bounds__(128) void k_lstm(const half_t* __restrict__ whh1,
                                              const half_t* __restrict__ wih2,
                                              const half_t* __restrict__ whh2,
                                              const float* __restrict__ g1x,
                                              const float* __restrict__ b2,
                                              const float* __restrict__ outw,
                                              half_t* __restrict__ h1g,   // [2][32][512]
                                              half_t* __restrict__ h2g,   // [2][32][512]
                                              float* __restrict__ opart,  // [T][32][16]
                                              unsigned* __restrict__ bar) {
    extern __shared__ char smem[];
    half_t* w1s = (half_t*)smem;                 // [4][32][512]
    half_t* w2s = w1s + 4 * 32 * 512;            // [4][32][512]
    float*  psum = (float*)(w2s + 4 * 32 * 512); // [2][32]

    const int tid = threadIdx.x;
    const int lane = tid & 31, wv = tid >> 5;
    const int sel = lane >> 4, l15 = lane & 15;
    const int g = blockIdx.x, hb = g * 32;
    const int mt = wv & 1, nblk = (wv >> 1) & 1;
    const int nloc = nblk * 16 + l15, nglob = hb + nloc;

    // One TDM descriptor per weight matrix: 3D tile x=512 halves, y=32 rows
    // (stride 512), z=4 gate blocks (stride 512*512). Wave 0 issues both DMAs
    // (TDM ignores EXEC; the uniform branch keeps other waves from re-issuing),
    // waits on TENSORcnt, then the workgroup barrier publishes the LDS slab.
    if (wv == 0) {
        tdm_load_slab(whh1 + (size_t)hb * 512, w1s, 512u, 512u, 512u * 512u);
        tdm_load_slab(whh2 + (size_t)hb * 512, w2s, 512u, 512u, 512u * 512u);
        __builtin_amdgcn_s_wait_tensorcnt(0);
    }
    float bv2[4];
#pragma unroll
    for (int G = 0; G < 4; ++G) bv2[G] = b2[G * 512 + nglob];
    const float owl = outw[nglob];

    v8f c1 = zero8(), h1 = zero8(), c2 = zero8(), h2 = zero8();
    __syncthreads();

#pragma unroll 1
    for (int t = 0; t < 1000; ++t) {
        const int rp = t & 1, wp = rp ^ 1;
        v8f acc[4];

        // ---- layer 1: gates = g1x[t] + h1_prev @ w_hh1^T ----
#pragma unroll
        for (int G = 0; G < 4; ++G) {
            v8f c;
#pragma unroll
            for (int r = 0; r < 8; ++r) {
                int m = mt * 16 + sel * 8 + r;
                c[r] = g1x[((size_t)(t * 32 + m)) * 2048 + G * 512 + nglob];
            }
            acc[G] = c;
        }
        for (int kk = 0; kk < 512; kk += 32) {
            v16h a = load_afrag(h1g + (size_t)rp * 32 * 512, 512, mt * 16 + l15, kk, sel);
#pragma unroll
            for (int G = 0; G < 4; ++G)
                acc[G] = wmma16(a, load_bfrag(w1s + (size_t)G * 32 * 512, 512, nloc, kk, sel), acc[G]);
        }
#pragma unroll
        for (int r = 0; r < 8; ++r) {
            float ig = sigm(acc[0][r]), fg = sigm(acc[1][r]);
            float gg = tanhf(acc[2][r]), og = sigm(acc[3][r]);
            float cn = fg * c1[r] + ig * gg;
            float hn = og * tanhf(cn);
            c1[r] = 0.1f * c1[r] + 0.9f * cn;        // zoneout (eval blend)
            h1[r] = 0.1f * h1[r] + 0.9f * hn;
            int m = mt * 16 + sel * 8 + r;
            h1g[(size_t)wp * 32 * 512 + (size_t)m * 512 + nglob] = (half_t)h1[r];
        }
        grid_barrier(bar, 16, tid);

        // ---- layer 2: gates = b2 + h1_new @ w_ih2^T + h2_prev @ w_hh2^T ----
#pragma unroll
        for (int G = 0; G < 4; ++G) {
            v8f c;
#pragma unroll
            for (int r = 0; r < 8; ++r) c[r] = bv2[G];
            acc[G] = c;
        }
        for (int kk = 0; kk < 512; kk += 32) {
            v16h a = load_afrag(h1g + (size_t)wp * 32 * 512, 512, mt * 16 + l15, kk, sel);
#pragma unroll
            for (int G = 0; G < 4; ++G)
                acc[G] = wmma16(a, load_bfrag(wih2, 512, G * 512 + nglob, kk, sel), acc[G]);
        }
        for (int kk = 0; kk < 512; kk += 32) {
            v16h a = load_afrag(h2g + (size_t)rp * 32 * 512, 512, mt * 16 + l15, kk, sel);
#pragma unroll
            for (int G = 0; G < 4; ++G)
                acc[G] = wmma16(a, load_bfrag(w2s + (size_t)G * 32 * 512, 512, nloc, kk, sel), acc[G]);
        }
        float prt[8];
#pragma unroll
        for (int r = 0; r < 8; ++r) {
            float ig = sigm(acc[0][r]), fg = sigm(acc[1][r]);
            float gg = tanhf(acc[2][r]), og = sigm(acc[3][r]);
            float cn = fg * c2[r] + ig * gg;
            float hn = og * tanhf(cn);
            c2[r] = 0.1f * c2[r] + 0.9f * cn;
            h2[r] = 0.1f * h2[r] + 0.9f * hn;
            int m = mt * 16 + sel * 8 + r;
            h2g[(size_t)wp * 32 * 512 + (size_t)m * 512 + nglob] = (half_t)h2[r];
            prt[r] = h2[r] * owl;                   // output-projection partial
        }
#pragma unroll
        for (int r = 0; r < 8; ++r)
            for (int mk = 1; mk < 16; mk <<= 1) prt[r] += __shfl_xor(prt[r], mk, 32);
        if (l15 == 0) {
#pragma unroll
            for (int r = 0; r < 8; ++r) psum[nblk * 32 + mt * 16 + sel * 8 + r] = prt[r];
        }
        __syncthreads();
        if (tid < 32) opart[((size_t)t * 32 + tid) * 16 + g] = psum[tid] + psum[32 + tid];
        grid_barrier(bar, 16, tid);
    }
}

// ---------------------------------------------------------------- final projection
__global__ __launch_bounds__(256) void k_final(const float* __restrict__ encf,
                                               const float* __restrict__ opart,
                                               const float* __restrict__ outw,
                                               const float* __restrict__ outb,
                                               const unsigned char* __restrict__ mask,
                                               float* __restrict__ out) {
    const int wv = threadIdx.x >> 5, lane = threadIdx.x & 31;
    const int row = blockIdx.x * 8 + wv;       // b*1000+t
    const int b = row / 1000, t = row - b * 1000;
    float s = 0.f;
#pragma unroll
    for (int j = 0; j < 16; ++j) {
        int c = lane + j * 32;
        s += encf[(size_t)row * 512 + c] * outw[512 + c];
    }
    if (lane < 16) s += opart[((size_t)t * 32 + b) * 16 + lane];
    for (int mk = 1; mk < 32; mk <<= 1) s += __shfl_xor(s, mk, 32);
    if (lane == 0) out[row] = mask[row] ? 0.f : (s + outb[0]);
}

// ---------------------------------------------------------------- launch
extern "C" void kernel_launch(void* const* d_in, const int* in_sizes, int n_in,
                              void* d_out, int out_size, void* d_ws, size_t ws_size,
                              hipStream_t stream) {
    const float* enc_in  = (const float*)d_in[0];
    const unsigned char* mask = (const unsigned char*)d_in[1];
    const float* target  = (const float*)d_in[2];
    const float* conv1_w = (const float*)d_in[3];
    const float* conv1_b = (const float*)d_in[4];
    const float* ln1_g   = (const float*)d_in[5];
    const float* ln1_b   = (const float*)d_in[6];
    const float* conv2_w = (const float*)d_in[7];
    const float* conv2_b = (const float*)d_in[8];
    const float* ln2_g   = (const float*)d_in[9];
    const float* ln2_b   = (const float*)d_in[10];
    const float* w_ih1   = (const float*)d_in[11];
    const float* w_hh1   = (const float*)d_in[12];
    const float* b_ih1   = (const float*)d_in[13];
    const float* b_hh1   = (const float*)d_in[14];
    const float* w_ih2   = (const float*)d_in[15];
    const float* w_hh2   = (const float*)d_in[16];
    const float* b_ih2   = (const float*)d_in[17];
    const float* b_hh2   = (const float*)d_in[18];
    const float* pren_w  = (const float*)d_in[19];
    const float* pren_b  = (const float*)d_in[20];
    const float* out_w   = (const float*)d_in[21];
    const float* out_b   = (const float*)d_in[22];
    float* out = (float*)d_out;

    char* ws = (char*)d_ws;
    size_t off = 0;
    auto take = [&](size_t bytes) { size_t o = off; off = (off + bytes + 255) & ~(size_t)255; return o; };
    unsigned* bar    = (unsigned*)(ws + take(256));
    float*  b1       = (float*) (ws + take(2048 * 4));
    float*  b2       = (float*) (ws + take(2048 * 4));
    half_t* wih1h    = (half_t*)(ws + take((size_t)2048 * 1024 * 2));
    half_t* whh1h    = (half_t*)(ws + take((size_t)2048 * 512 * 2));
    half_t* wih2h    = (half_t*)(ws + take((size_t)2048 * 512 * 2));
    half_t* whh2h    = (half_t*)(ws + take((size_t)2048 * 512 * 2));
    half_t* w1t      = (half_t*)(ws + take((size_t)512 * 1152 * 2));
    half_t* w2t      = (half_t*)(ws + take((size_t)512 * 1536 * 2));
    half_t* enc16    = (half_t*)(ws + take((size_t)32000 * 384 * 2));
    half_t* x1h      = (half_t*)(ws + take((size_t)32000 * 512 * 2));
    float*  ytmp     = (float*) (ws + take((size_t)32000 * 512 * 4));
    float*  encf     = (float*) (ws + take((size_t)32000 * 512 * 4));
    half_t* xs       = (half_t*)(ws + take((size_t)32000 * 1024 * 2));
    float*  g1x      = (float*) (ws + take((size_t)32000 * 2048 * 4));
    half_t* h1g      = (half_t*)(ws + take((size_t)2 * 32 * 512 * 2));
    half_t* h2g      = (half_t*)(ws + take((size_t)2 * 32 * 512 * 2));
    float*  opart    = (float*) (ws + take((size_t)1000 * 32 * 16 * 4));

    auto cdiv = [](int a, int b) { return (a + b - 1) / b; };

    // -- prep: f16 weight staging, fused biases, conv-weight repack, state zeroing
    k_f32_to_f16<<<cdiv(2048 * 1024, 256), 256, 0, stream>>>(w_ih1, wih1h, 2048 * 1024);
    k_f32_to_f16<<<cdiv(2048 * 512, 256), 256, 0, stream>>>(w_hh1, whh1h, 2048 * 512);
    k_f32_to_f16<<<cdiv(2048 * 512, 256), 256, 0, stream>>>(w_ih2, wih2h, 2048 * 512);
    k_f32_to_f16<<<cdiv(2048 * 512, 256), 256, 0, stream>>>(w_hh2, whh2h, 2048 * 512);
    k_f32_to_f16<<<cdiv(32000 * 384, 256), 256, 0, stream>>>(enc_in, enc16, 32000 * 384);
    k_bias_sum<<<cdiv(2048, 256), 256, 0, stream>>>(b_ih1, b_hh1, b1, 2048);
    k_bias_sum<<<cdiv(2048, 256), 256, 0, stream>>>(b_ih2, b_hh2, b2, 2048);
    k_repack_convw<<<cdiv(512 * 384 * 3, 256), 256, 0, stream>>>(conv1_w, w1t, 512, 384);
    k_repack_convw<<<cdiv(512 * 512 * 3, 256), 256, 0, stream>>>(conv2_w, w2t, 512, 512);
    k_zero_u32<<<cdiv(16384, 256), 256, 0, stream>>>((unsigned*)h1g, 16384);
    k_zero_u32<<<cdiv(16384, 256), 256, 0, stream>>>((unsigned*)h2g, 16384);
    k_zero_u32<<<1, 64, 0, stream>>>(bar, 2);
    k_prenet<<<cdiv(32000 * 512, 256), 256, 0, stream>>>(target, pren_w, pren_b, xs);

    // -- conv prenet stack (WMMA implicit-im2col GEMMs + relu/LN)
    k_conv_wmma<<<2000, 256, 0, stream>>>(enc16, w1t, conv1_b, ytmp, 384);
    k_ln<0><<<4000, 256, 0, stream>>>(ytmp, ln1_g, ln1_b, x1h, nullptr, nullptr);
    k_conv_wmma<<<2000, 256, 0, stream>>>(x1h, w2t, conv2_b, ytmp, 512);
    k_ln<1><<<4000, 256, 0, stream>>>(ytmp, ln2_g, ln2_b, nullptr, encf, xs);

    // -- hoisted input-gate GEMM for layer 1 over all timesteps
    k_gemm_ih1<<<dim3(2000, 4), 256, 0, stream>>>(xs, wih1h, b1, g1x);

    // -- persistent recurrent kernel (16 WGs, TDM-staged LDS-pinned weights)
    size_t lstm_lds = (size_t)2 * 4 * 32 * 512 * sizeof(half_t) + 64 * sizeof(float);
    k_lstm<<<16, 128, lstm_lds, stream>>>(whh1h, wih2h, whh2h, g1x, b2, out_w,
                                          h1g, h2g, opart, bar);

    // -- output projection epilogue
    k_final<<<4000, 256, 0, stream>>>(encf, opart, out_w, out_b, mask, out);
}
